// CGCNN_1271310320289
// MI455X (gfx1250) — matrix operand order repaired and checked by reading, not compiled, not run on hardware
//
#include <hip/hip_runtime.h>

// ---------------------------------------------------------------------------
// CGCNN on MI455X (gfx1250): edge-gathered GEMMs via v_wmma_f32_16x16x32_bf16.
// M-blocked: each wave handles 32 edges (2 M-tiles), reusing every weight
// (B) fragment for two WMMAs.
// ---------------------------------------------------------------------------

typedef __attribute__((ext_vector_type(16))) __bf16 v16bf;
typedef __attribute__((ext_vector_type(8)))  float  v8f;

#define N_NODES     50000
#define N_EDGES     1600000
#define IN_CH       64
#define EDGE_DIM    32
#define ZDIM        160          // 2*IN_CH + EDGE_DIM = 5 * 32 K-steps
#define NUM_HID     256
#define NUM_CLASSES 16
#define NUM_GRAPHS  512

// convert float4 -> 4 bf16, store as one 8-byte LDS write
__device__ __forceinline__ void st_bf16x4(__bf16* p, float4 v) {
    __bf16 b0 = (__bf16)v.x, b1 = (__bf16)v.y, b2 = (__bf16)v.z, b3 = (__bf16)v.w;
    unsigned short s0, s1, s2, s3;
    __builtin_memcpy(&s0, &b0, 2); __builtin_memcpy(&s1, &b1, 2);
    __builtin_memcpy(&s2, &b2, 2); __builtin_memcpy(&s3, &b3, 2);
    uint2 u;
    u.x = (unsigned)s0 | ((unsigned)s1 << 16);
    u.y = (unsigned)s2 | ((unsigned)s3 << 16);
    *(uint2*)p = u;
}

// ---------------------------------------------------------------------------
// small helpers
// ---------------------------------------------------------------------------
__global__ void f2bf_kernel(const float* __restrict__ in, __bf16* __restrict__ out, int n) {
    int i = blockIdx.x * blockDim.x + threadIdx.x;
    if (i < n) out[i] = (__bf16)in[i];
}

__global__ void deg_count_kernel(const int* __restrict__ dst, float* __restrict__ deg) {
    int i = blockIdx.x * blockDim.x + threadIdx.x;
    if (i < N_EDGES) atomicAdd(&deg[dst[i]], 1.0f);
}

__global__ void deg_inv_kernel(float* __restrict__ deg) {
    int i = blockIdx.x * blockDim.x + threadIdx.x;
    if (i < N_NODES) deg[i] = 1.0f / fmaxf(deg[i], 1.0f);
}

// h = agg * inv_deg + x_in   (residual)
__global__ void node_update_kernel(const float* __restrict__ agg,
                                   const float* __restrict__ invdeg,
                                   const float* __restrict__ xin,
                                   float* __restrict__ hout) {
    int i = blockIdx.x * blockDim.x + threadIdx.x;
    if (i < N_NODES * IN_CH) {
        int node = i >> 6;
        hout[i] = agg[i] * invdeg[node] + xin[i];
    }
}

// ---------------------------------------------------------------------------
// CGConv edge kernel: per wave, one 32-edge tile (2 WMMA M-tiles).
//   z(32x160) bf16 in LDS;  msg = sigmoid(z Wf^T + bf) * softplus(z Ws^T + bs)
//   scatter-add msg into agg[dst].
// ---------------------------------------------------------------------------
__launch_bounds__(128)
__global__ void cgconv_edge_kernel(const float* __restrict__ xin,
                                   const int*   __restrict__ src,
                                   const int*   __restrict__ dst,
                                   const float* __restrict__ eattr,
                                   const __bf16* __restrict__ wf,   // [64][160] bf16
                                   const __bf16* __restrict__ wsm,  // [64][160] bf16
                                   const float* __restrict__ bf,    // [64]
                                   const float* __restrict__ bs,    // [64]
                                   float* __restrict__ agg)         // [N_NODES][64]
{
    __shared__ __bf16 zt[4][32][ZDIM];          // 40 KB: one 32x160 tile per wave

    const int wave  = threadIdx.x >> 5;
    const int lane  = threadIdx.x & 31;
    const int tile0 = (blockIdx.x * 4 + wave) * 32;   // first edge of this tile

    // ---- stage z tile into LDS (bf16): each lane stages one full edge row --
    {
        const int e = tile0 + lane;
        const int s = src[e];
        const int d = dst[e];
        const float4* xd = (const float4*)(xin + (size_t)d * IN_CH);
        const float4* xs = (const float4*)(xin + (size_t)s * IN_CH);
        const float4* ep = (const float4*)(eattr + (size_t)e * EDGE_DIM);
        __bf16* zr = zt[wave][lane];

        #pragma unroll
        for (int q = 0; q < 16; ++q) st_bf16x4(zr + q * 4, xd[q]);          // x[dst]
        #pragma unroll
        for (int q = 0; q < 16; ++q) st_bf16x4(zr + 64 + q * 4, xs[q]);     // x[src]
        #pragma unroll
        for (int q = 0; q < 8;  ++q) st_bf16x4(zr + 128 + q * 4, ep[q]);    // edge_attr
    }
    __syncthreads();

    // ---- 32x160 @ 160x64 twice (gate + filter), bf16 WMMA, f32 acc --------
    v8f accf[2][4];
    v8f accs[2][4];
    #pragma unroll
    for (int t = 0; t < 2; ++t)
        #pragma unroll
        for (int nt = 0; nt < 4; ++nt) {
            accf[t][nt] = (v8f){0.f,0.f,0.f,0.f,0.f,0.f,0.f,0.f};
            accs[t][nt] = (v8f){0.f,0.f,0.f,0.f,0.f,0.f,0.f,0.f};
        }

    const int m     = lane & 15;               // A row-in-tile / B column
    const int koffA = (lane >> 4) << 3;        // A-layout: 0 or 8
    const int koffB = (lane >> 4) << 4;        // B-layout: 0 or 16

    #pragma unroll
    for (int ks = 0; ks < 5; ++ks) {
        const int kbase = ks * 32;

        // two A fragments (rows 0..15 and 16..31), contiguous k = 16*(lane/16) + s, same order as B
        v16bf a0, a1;
        #pragma unroll
        for (int j = 0; j < 8; ++j) {
            a0[j]     = zt[wave][m][kbase + koffB + j];
            a0[8 + j] = zt[wave][m][kbase + koffB + 8 + j];
            a1[j]     = zt[wave][16 + m][kbase + koffB + j];
            a1[8 + j] = zt[wave][16 + m][kbase + koffB + 8 + j];
        }

        #pragma unroll
        for (int nt = 0; nt < 4; ++nt) {
            const int n = nt * 16 + m;         // output channel = row of W
            v16bf bfv = *(const v16bf*)(wf  + (size_t)n * ZDIM + kbase + koffB);
            v16bf bsv = *(const v16bf*)(wsm + (size_t)n * ZDIM + kbase + koffB);
            accf[0][nt] = __builtin_amdgcn_wmma_f32_16x16x32_bf16(
                false, a0, false, bfv, (short)0, accf[0][nt], false, false);
            accf[1][nt] = __builtin_amdgcn_wmma_f32_16x16x32_bf16(
                false, a1, false, bfv, (short)0, accf[1][nt], false, false);
            accs[0][nt] = __builtin_amdgcn_wmma_f32_16x16x32_bf16(
                false, a0, false, bsv, (short)0, accs[0][nt], false, false);
            accs[1][nt] = __builtin_amdgcn_wmma_f32_16x16x32_bf16(
                false, a1, false, bsv, (short)0, accs[1][nt], false, false);
        }
    }

    // ---- epilogue: bias, sigmoid*softplus, atomic scatter to agg[dst] -----
    const int mrow = (lane >> 4) << 3;         // C/D layout: rows r+0 or r+8
    int drow[2][8];
    #pragma unroll
    for (int t = 0; t < 2; ++t)
        #pragma unroll
        for (int r = 0; r < 8; ++r)
            drow[t][r] = dst[tile0 + t * 16 + mrow + r];

    #pragma unroll
    for (int nt = 0; nt < 4; ++nt) {
        const int   ch  = nt * 16 + m;
        const float bfb = bf[ch];
        const float bsb = bs[ch];
        #pragma unroll
        for (int t = 0; t < 2; ++t) {
            #pragma unroll
            for (int r = 0; r < 8; ++r) {
                float g  = accf[t][nt][r] + bfb;
                float sg = 1.0f / (1.0f + __expf(-g));
                float sv = accs[t][nt][r] + bsb;
                float sp = (sv > 15.0f) ? sv : __logf(1.0f + __expf(sv));
                atomicAdd(&agg[(size_t)drow[t][r] * IN_CH + ch], sg * sp);
            }
        }
    }
}

// ---------------------------------------------------------------------------
// graph pooling (scatter-mean) + MLP head
// ---------------------------------------------------------------------------
__global__ void pool_sum_kernel(const float* __restrict__ h, const int* __restrict__ batch,
                                float* __restrict__ gsum) {
    int i = blockIdx.x * blockDim.x + threadIdx.x;
    if (i < N_NODES * IN_CH) {
        int node = i >> 6, c = i & 63;
        atomicAdd(&gsum[(size_t)batch[node] * IN_CH + c], h[i]);
    }
}

__global__ void pool_cnt_kernel(const int* __restrict__ batch, float* __restrict__ gcnt) {
    int i = blockIdx.x * blockDim.x + threadIdx.x;
    if (i < N_NODES) atomicAdd(&gcnt[batch[i]], 1.0f);
}

__launch_bounds__(256)
__global__ void mlp_kernel(const float* __restrict__ gsum, const float* __restrict__ gcnt,
                           const float* __restrict__ w1, const float* __restrict__ b1,
                           const float* __restrict__ w2, const float* __restrict__ b2,
                           float* __restrict__ out) {
    __shared__ float gv[IN_CH];
    __shared__ float hid[NUM_HID];
    __shared__ float o[NUM_CLASSES];
    __shared__ float red[2];

    const int g = blockIdx.x;
    const int t = threadIdx.x;
    const float inv = 1.0f / fmaxf(gcnt[g], 1.0f);
    if (t < IN_CH) gv[t] = gsum[(size_t)g * IN_CH + t] * inv;
    __syncthreads();

    float acc = b1[t];
    #pragma unroll 8
    for (int c = 0; c < IN_CH; ++c) acc += gv[c] * w1[(size_t)t * IN_CH + c];
    hid[t] = fmaxf(acc, 0.0f);
    __syncthreads();

    if (t < NUM_CLASSES) {
        float a2 = b2[t];
        #pragma unroll 8
        for (int j = 0; j < NUM_HID; ++j) a2 += hid[j] * w2[(size_t)t * NUM_HID + j];
        o[t] = a2;
    }
    __syncthreads();

    if (t == 0) {
        float mx = o[0];
        for (int k = 1; k < NUM_CLASSES; ++k) mx = fmaxf(mx, o[k]);
        float s = 0.0f;
        for (int k = 0; k < NUM_CLASSES; ++k) s += __expf(o[k] - mx);
        red[0] = mx;
        red[1] = logf(s);
    }
    __syncthreads();

    if (t < NUM_CLASSES) out[(size_t)g * NUM_CLASSES + t] = o[t] - red[0] - red[1];
}

// ---------------------------------------------------------------------------
// launch
// ---------------------------------------------------------------------------
extern "C" void kernel_launch(void* const* d_in, const int* in_sizes, int n_in,
                              void* d_out, int out_size, void* d_ws, size_t ws_size,
                              hipStream_t stream) {
    (void)in_sizes; (void)n_in; (void)out_size; (void)ws_size;

    const float* x     = (const float*)d_in[0];
    const int*   ei    = (const int*)d_in[1];
    const int*   src   = ei;
    const int*   dst   = ei + N_EDGES;
    const float* ea    = (const float*)d_in[2];
    const int*   batch = (const int*)d_in[3];
    const float* wf1 = (const float*)d_in[4];  const float* bf1 = (const float*)d_in[5];
    const float* ws1 = (const float*)d_in[6];  const float* bs1 = (const float*)d_in[7];
    const float* wf2 = (const float*)d_in[8];  const float* bf2 = (const float*)d_in[9];
    const float* ws2 = (const float*)d_in[10]; const float* bs2 = (const float*)d_in[11];
    const float* w1  = (const float*)d_in[12]; const float* b1  = (const float*)d_in[13];
    const float* w2  = (const float*)d_in[14]; const float* b2  = (const float*)d_in[15];

    // workspace layout
    char* ws = (char*)d_ws;
    size_t off = 0;
    auto alloc = [&](size_t bytes) -> void* {
        void* p = ws + off;
        off = (off + bytes + 255) & ~(size_t)255;
        return p;
    };
    float*  deg  = (float*)alloc((size_t)N_NODES * 4);
    float*  agg  = (float*)alloc((size_t)N_NODES * IN_CH * 4);
    float*  h1   = (float*)alloc((size_t)N_NODES * IN_CH * 4);
    float*  h2   = (float*)alloc((size_t)N_NODES * IN_CH * 4);
    __bf16* wfb1 = (__bf16*)alloc((size_t)IN_CH * ZDIM * 2);
    __bf16* wsb1 = (__bf16*)alloc((size_t)IN_CH * ZDIM * 2);
    __bf16* wfb2 = (__bf16*)alloc((size_t)IN_CH * ZDIM * 2);
    __bf16* wsb2 = (__bf16*)alloc((size_t)IN_CH * ZDIM * 2);
    float*  gsum = (float*)alloc((size_t)NUM_GRAPHS * IN_CH * 4);
    float*  gcnt = (float*)alloc((size_t)NUM_GRAPHS * 4);

    const int WN = IN_CH * ZDIM;  // 10240 weight elements per matrix

    // weights -> bf16 (tiny, cache resident afterwards)
    f2bf_kernel<<<(WN + 255) / 256, 256, 0, stream>>>(wf1, wfb1, WN);
    f2bf_kernel<<<(WN + 255) / 256, 256, 0, stream>>>(ws1, wsb1, WN);
    f2bf_kernel<<<(WN + 255) / 256, 256, 0, stream>>>(wf2, wfb2, WN);
    f2bf_kernel<<<(WN + 255) / 256, 256, 0, stream>>>(ws2, wsb2, WN);

    // degree -> inv_deg
    hipMemsetAsync(deg, 0, (size_t)N_NODES * 4, stream);
    deg_count_kernel<<<(N_EDGES + 255) / 256, 256, 0, stream>>>(dst, deg);
    deg_inv_kernel<<<(N_NODES + 255) / 256, 256, 0, stream>>>(deg);

    const int edgeBlocks = N_EDGES / 128;                // 4 waves x 32 edges
    const int nodeElems  = N_NODES * IN_CH;

    // ---- CGConv layer 1 ----
    hipMemsetAsync(agg, 0, (size_t)nodeElems * 4, stream);
    cgconv_edge_kernel<<<edgeBlocks, 128, 0, stream>>>(x, src, dst, ea,
                                                       wfb1, wsb1, bf1, bs1, agg);
    node_update_kernel<<<(nodeElems + 255) / 256, 256, 0, stream>>>(agg, deg, x, h1);

    // ---- CGConv layer 2 ----
    hipMemsetAsync(agg, 0, (size_t)nodeElems * 4, stream);
    cgconv_edge_kernel<<<edgeBlocks, 128, 0, stream>>>(h1, src, dst, ea,
                                                       wfb2, wsb2, bf2, bs2, agg);
    node_update_kernel<<<(nodeElems + 255) / 256, 256, 0, stream>>>(agg, deg, h1, h2);

    // ---- pooling + MLP head ----
    hipMemsetAsync(gsum, 0, (size_t)NUM_GRAPHS * IN_CH * 4, stream);
    hipMemsetAsync(gcnt, 0, (size_t)NUM_GRAPHS * 4, stream);
    pool_sum_kernel<<<(nodeElems + 255) / 256, 256, 0, stream>>>(h2, batch, gsum);
    pool_cnt_kernel<<<(N_NODES + 255) / 256, 256, 0, stream>>>(batch, gcnt);
    mlp_kernel<<<NUM_GRAPHS, 256, 0, stream>>>(gsum, gcnt, w1, b1, w2, b2, (float*)d_out);
}
